// Mutihead_Attention_36833639530818
// MI455X (gfx1250) — matrix operand order repaired and checked by reading, not compiled, not run on hardware
//
#include <hip/hip_runtime.h>
#include <hip/hip_bf16.h>
#include <math.h>

// MI455X (gfx1250) fused multi-head attention.
// All matmuls: v_wmma_f32_16x16x32_f16 (f32 accumulation).
// Attention uses a shift-by-row-mean softmax (shift-invariant, no online max),
// so the inner loop is pure b128 loads + WMMA + exp. Workspace: ~53 MB.

typedef __attribute__((ext_vector_type(16))) _Float16 v16h;
typedef __attribute__((ext_vector_type(8)))  _Float16 v8h;
typedef __attribute__((ext_vector_type(4)))  _Float16 v4h;
typedef __attribute__((ext_vector_type(8)))  float    v8f;
typedef __attribute__((ext_vector_type(4)))  float    v4f;

#define NORM_SCALE 0.04419417382415922f   // 1/sqrt(512)

__device__ __forceinline__ v8f wmma_f16(v16h a, v16h b, v8f c) {
  // D = A(16x32,f16) * B(32x16,f16) + C(16x16,f32)
  return __builtin_amdgcn_wmma_f32_16x16x32_f16(false, a, false, b, (short)0, c,
                                                false, false);
}

// A-matrix fragment (16x32) from row-major f16 (ld halfs).
// Lane(0-15)=M row; element j -> K = (j%8) + 16*(j/8) + 8*(lane>=16).
__device__ __forceinline__ v16h load_a(const _Float16* __restrict__ M, int ld,
                                       int m0, int k0) {
  const int lane = threadIdx.x & 31;
  const _Float16* p = M + (m0 + (lane & 15)) * ld + k0 + ((lane >> 4) << 3);
  v8h lo = *reinterpret_cast<const v8h*>(p);
  v8h hi = *reinterpret_cast<const v8h*>(p + 16);
  v16h f;
#pragma unroll
  for (int j = 0; j < 8; ++j) { f[j] = lo[j]; f[8 + j] = hi[j]; }
  return f;
}

// B-matrix fragment (32x16) where memory holds Bt[n][k] f16 row-major
// (contraction contiguous per lane). Lane(0-15)=N; element j -> K = k0 +
// (j%8) + 16*(j/8) + 8*(lane>=16).
__device__ __forceinline__ v16h load_bt(const _Float16* __restrict__ Bt, int ld,
                                        int k0, int n0) {
  const int lane = threadIdx.x & 31;
  const _Float16* p = Bt + (n0 + (lane & 15)) * ld + k0 + ((lane >> 4) << 3);
  v8h lo = *reinterpret_cast<const v8h*>(p);
  v8h hi = *reinterpret_cast<const v8h*>(p + 16);
  v16h f;
#pragma unroll
  for (int j = 0; j < 8; ++j) { f[j] = lo[j]; f[8 + j] = hi[j]; }
  return f;
}

// ---------------------------------------------------------------------------
// Elementwise f32 -> f16 (for x, y). 4 elements/thread.
// ---------------------------------------------------------------------------
__global__ void __launch_bounds__(256) cvt_f16_kernel(
    const float* __restrict__ in, _Float16* __restrict__ out) {
  const int i = (blockIdx.x * 256 + threadIdx.x) << 2;
  v4f v = *reinterpret_cast<const v4f*>(in + i);
  v4h h;
#pragma unroll
  for (int j = 0; j < 4; ++j) h[j] = (_Float16)v[j];
  *reinterpret_cast<v4h*>(out + i) = h;
}

// ---------------------------------------------------------------------------
// Weight transpose+convert: Wt[n][k] = (f16) W[k][n], 512x512.
// ---------------------------------------------------------------------------
__global__ void __launch_bounds__(256) wtrans_kernel(
    const float* __restrict__ W, _Float16* __restrict__ Wt) {
  __shared__ float tile[16][17];
  const int tx = threadIdx.x & 15, ty = threadIdx.x >> 4;
  const int k = (blockIdx.y << 4) + ty;
  const int n = (blockIdx.x << 4) + tx;
  tile[ty][tx] = W[k * 512 + n];
  __syncthreads();
  const int no = (blockIdx.x << 4) + ty;
  const int ko = (blockIdx.y << 4) + tx;
  Wt[no * 512 + ko] = (_Float16)tile[tx][ty];
}

// ---------------------------------------------------------------------------
// GEMM+bias: C[8192,512] = A[8192,512](f16) @ Wt^T (Wt is [N][K] f16) + bias
// Block = 128 threads (4 waves). Wave computes a 32x64 tile (8 f32 accs):
// per k-step: 4+8 b128 loads, 8 WMMAs. Grid: (8, 64).
// MODE 0: f32 out, flat [row][col]     (final projection -> d_out)
// MODE 1: f16 out, flat [row][col]     (Q, K)
// MODE 2: f16 out, V-transposed: Vt[hb][dv][key] with hb=row>>7,
//         key=((row&127)<<3)|(col>>6), dv=col&63   (V)
// ---------------------------------------------------------------------------
template <int MODE>
__global__ void __launch_bounds__(128) gemm512(
    const _Float16* __restrict__ A, const _Float16* __restrict__ Wt,
    const float* __restrict__ bias, float* __restrict__ Cf,
    _Float16* __restrict__ Ch) {
  const int lane = threadIdx.x & 31;
  const int wave = threadIdx.x >> 5;
  const int n0 = blockIdx.x << 6;
  const int m0 = (blockIdx.y << 7) + (wave << 5);

  v8f acc[2][4] = {};
  for (int k0 = 0; k0 < 512; k0 += 32) {
    __builtin_prefetch(A + (m0 + (lane & 15)) * 512 + k0 + 32);
    __builtin_prefetch(Wt + (n0 + (lane & 15)) * 512 + k0 + 32);
    v16h a0 = load_a(A, 512, m0, k0);
    v16h a1 = load_a(A, 512, m0 + 16, k0);
#pragma unroll
    for (int t = 0; t < 4; ++t) {
      v16h b = load_bt(Wt, 512, k0, n0 + (t << 4));
      acc[0][t] = wmma_f16(a0, b, acc[0][t]);
      acc[1][t] = wmma_f16(a1, b, acc[1][t]);
    }
  }

  const int nn = lane & 15;
  const int rbase = (lane >> 4) << 3;
#pragma unroll
  for (int u = 0; u < 2; ++u) {
#pragma unroll
    for (int t = 0; t < 4; ++t) {
      const int col = n0 + (t << 4) + nn;
      const float bv = bias[col];
#pragma unroll
      for (int r = 0; r < 8; ++r) {
        const int row = m0 + (u << 4) + rbase + r;
        const float v = acc[u][t][r] + bv;
        if (MODE == 0) {
          Cf[row * 512 + col] = v;
        } else if (MODE == 1) {
          Ch[row * 512 + col] = (_Float16)v;
        } else {
          const int hb = row >> 7;
          const int key = ((row & 127) << 3) | (col >> 6);
          const int dv = col & 63;
          Ch[hb * 65536 + dv * 1024 + key] = (_Float16)v;
        }
      }
    }
  }
}

// ---------------------------------------------------------------------------
// Ksum[hb][d] = sum_s K[hb][s][d]   (Kf viewed as [64][1024][64] f16)
// ---------------------------------------------------------------------------
__global__ void __launch_bounds__(256) ksum_kernel(
    const _Float16* __restrict__ K, float* __restrict__ Ksum) {
  __shared__ float red[256];
  const int hb = blockIdx.x;
  const int d  = threadIdx.x & 63;
  const int g  = threadIdx.x >> 6;
  const _Float16* p = K + hb * 65536 + d;
  float s = 0.0f;
  for (int t = g; t < 1024; t += 4) s += (float)p[t * 64];
  red[threadIdx.x] = s;
  __syncthreads();
  if (g == 0)
    Ksum[hb * 64 + d] = red[d] + red[d + 64] + red[d + 128] + red[d + 192];
}

// ---------------------------------------------------------------------------
// thre[hb][q] = NORM * dot(Q[hb][q], Ksum[hb]) / 1024  (row mean of scores)
// ---------------------------------------------------------------------------
__global__ void __launch_bounds__(256) thre_kernel(
    const _Float16* __restrict__ Q, const float* __restrict__ Ksum,
    float* __restrict__ Thre) {
  const int hb = blockIdx.y;
  const int q  = blockIdx.x * 256 + threadIdx.x;
  const _Float16* qp = Q + hb * 65536 + q * 64;
  const float* ks = Ksum + hb * 64;
  float s = 0.0f;
#pragma unroll 8
  for (int d = 0; d < 64; ++d) s += (float)qp[d] * ks[d];
  Thre[hb * 1024 + q] = s * NORM_SCALE * (1.0f / 1024.0f);
}

// ---------------------------------------------------------------------------
// Fused attention with mean-threshold mask, shift-by-mean softmax:
//   p = (s > thre) ? exp(s - thre) : 0 ;  out = (P @ V) / rowsum(P)
// (softmax is shift-invariant; thre is the row mean, so exponents stay small)
// Block = 128 threads (4 waves), wave handles 32 q rows. Grid: (8, 64).
// Per 32-key tile: 4 K-frag loads -> 8 score WMMAs -> mask/exp ->
// LDS P-transpose -> 8 P@V WMMAs (V frags shared across both row tiles).
// ---------------------------------------------------------------------------
__global__ void __launch_bounds__(128) attn_kernel(
    const _Float16* __restrict__ Q, const _Float16* __restrict__ Km,
    const _Float16* __restrict__ Vt, const float* __restrict__ Thre,
    _Float16* __restrict__ O) {
  __shared__ _Float16 pbuf[4][2][16 * 32];  // per-wave P staging
  const int lane = threadIdx.x & 31;
  const int wave = threadIdx.x >> 5;
  const int hb   = blockIdx.y;
  const int q0   = (blockIdx.x * 4 + wave) << 5;  // 32 rows per wave

  const _Float16* Qb = Q  + hb * 65536;
  const _Float16* Kb = Km + hb * 65536;  // [key][d], d contiguous
  const _Float16* Vb = Vt + hb * 65536;  // [dv][key], key contiguous

  const int nn    = lane & 15;
  const int hi    = lane >> 4;
  const int rbase = hi << 3;

  v16h qa[2], qk[2];
  float trow[2][8], lpart[2][8];
  v8f oacc[2][4] = {};
#pragma unroll
  for (int u = 0; u < 2; ++u) {
    qa[u] = load_a(Qb, 64, q0 + (u << 4), 0);
    qk[u] = load_a(Qb, 64, q0 + (u << 4), 32);
#pragma unroll
    for (int r = 0; r < 8; ++r) {
      trow[u][r]  = Thre[hb * 1024 + q0 + (u << 4) + rbase + r];
      lpart[u][r] = 0.0f;
    }
  }

  for (int k0 = 0; k0 < 1024; k0 += 32) {
    __builtin_prefetch(Kb + (k0 + 32 + nn) * 64);
    __builtin_prefetch(Vb + nn * 1024 + k0 + 32);

    // score fragments for 32 keys (B = K^T, d-contiguous loads)
    const v16h b0a = load_bt(Kb, 64, 0,  k0);
    const v16h b0b = load_bt(Kb, 64, 32, k0);
    const v16h b1a = load_bt(Kb, 64, 0,  k0 + 16);
    const v16h b1b = load_bt(Kb, 64, 32, k0 + 16);

#pragma unroll
    for (int u = 0; u < 2; ++u) {
      v8f s1 = {}, s2 = {};
      s1 = wmma_f16(qa[u], b0a, s1);
      s1 = wmma_f16(qk[u], b0b, s1);
      s2 = wmma_f16(qa[u], b1a, s2);
      s2 = wmma_f16(qk[u], b1b, s2);

      _Float16* wp = &pbuf[wave][u][0];
#pragma unroll
      for (int r = 0; r < 8; ++r) {
        const float t = trow[u][r];
        const float a = s1[r] * NORM_SCALE;
        const float c = s2[r] * NORM_SCALE;
        const float pa = (a > t) ? __expf(a - t) : 0.0f;  // masked exp
        const float pc = (c > t) ? __expf(c - t) : 0.0f;
        lpart[u][r] += pa + pc;  // per-lane partial rowsum (fixed key subset)
        const int row = rbase + r;
        wp[row * 32 + nn]      = (_Float16)pa;
        wp[row * 32 + nn + 16] = (_Float16)pc;
      }
    }
    __syncthreads();

    // V fragments (key-contiguous), shared by both row tiles
    const v16h vb0 = load_bt(Vb, 1024, k0, 0);
    const v16h vb1 = load_bt(Vb, 1024, k0, 16);
    const v16h vb2 = load_bt(Vb, 1024, k0, 32);
    const v16h vb3 = load_bt(Vb, 1024, k0, 48);

#pragma unroll
    for (int u = 0; u < 2; ++u) {
      const _Float16* rp = &pbuf[wave][u][nn * 32 + (hi << 3)];
      v8h plo = *reinterpret_cast<const v8h*>(rp);
      v8h phi = *reinterpret_cast<const v8h*>(rp + 16);
      v16h pf;
#pragma unroll
      for (int j = 0; j < 8; ++j) { pf[j] = plo[j]; pf[8 + j] = phi[j]; }
      oacc[u][0] = wmma_f16(pf, vb0, oacc[u][0]);
      oacc[u][1] = wmma_f16(pf, vb1, oacc[u][1]);
      oacc[u][2] = wmma_f16(pf, vb2, oacc[u][2]);
      oacc[u][3] = wmma_f16(pf, vb3, oacc[u][3]);
    }
    __syncthreads();
  }

  // one-time rowsum reduction + normalize + f16 store
#pragma unroll
  for (int u = 0; u < 2; ++u) {
#pragma unroll
    for (int r = 0; r < 8; ++r) {
      float l = lpart[u][r];
      l += __shfl_xor(l, 1, 32);
      l += __shfl_xor(l, 2, 32);
      l += __shfl_xor(l, 4, 32);
      l += __shfl_xor(l, 8, 32);
      const float inv = (l > 0.0f) ? (1.0f / l) : 0.0f;
      _Float16* op = O + hb * 65536 + (q0 + (u << 4) + rbase + r) * 64;
#pragma unroll
      for (int t = 0; t < 4; ++t)
        op[(t << 4) + nn] = (_Float16)(oacc[u][t][r] * inv);
    }
  }
}

// ---------------------------------------------------------------------------
extern "C" void kernel_launch(void* const* d_in, const int* in_sizes, int n_in,
                              void* d_out, int out_size, void* d_ws,
                              size_t ws_size, hipStream_t stream) {
  (void)in_sizes; (void)n_in; (void)out_size; (void)ws_size;

  const float* x   = (const float*)d_in[0];
  const float* y   = (const float*)d_in[1];
  const float* q_w = (const float*)d_in[2];
  const float* q_b = (const float*)d_in[3];
  const float* k_w = (const float*)d_in[4];
  const float* k_b = (const float*)d_in[5];
  const float* v_w = (const float*)d_in[6];
  const float* v_b = (const float*)d_in[7];
  const float* o_w = (const float*)d_in[8];
  const float* o_b = (const float*)d_in[9];
  float* out = (float*)d_out;

  // workspace carve-up (all offsets well-aligned)
  const size_t NACT = 8ull * 1024 * 512;  // 4194304 elements
  char* ws = (char*)d_ws;
  _Float16* Xh  = (_Float16*)ws; ws += NACT * 2;
  _Float16* Yh  = (_Float16*)ws; ws += NACT * 2;
  _Float16* Qf  = (_Float16*)ws; ws += NACT * 2;
  _Float16* Kf  = (_Float16*)ws; ws += NACT * 2;
  _Float16* Vtr = (_Float16*)ws; ws += NACT * 2;
  _Float16* Of  = (_Float16*)ws; ws += NACT * 2;
  _Float16* Wtq = (_Float16*)ws; ws += 512 * 512 * 2;
  _Float16* Wtk = (_Float16*)ws; ws += 512 * 512 * 2;
  _Float16* Wtv = (_Float16*)ws; ws += 512 * 512 * 2;
  _Float16* Wto = (_Float16*)ws; ws += 512 * 512 * 2;
  float* Ksum = (float*)ws; ws += 4096 * 4;
  float* Thre = (float*)ws;

  // 1) activations -> f16
  cvt_f16_kernel<<<4096, 256, 0, stream>>>(x, Xh);
  cvt_f16_kernel<<<4096, 256, 0, stream>>>(y, Yh);
  // 2) weights -> f16, N-major (contraction contiguous)
  const dim3 wtG(32, 32);
  wtrans_kernel<<<wtG, 256, 0, stream>>>(q_w, Wtq);
  wtrans_kernel<<<wtG, 256, 0, stream>>>(k_w, Wtk);
  wtrans_kernel<<<wtG, 256, 0, stream>>>(v_w, Wtv);
  wtrans_kernel<<<wtG, 256, 0, stream>>>(o_w, Wto);

  // 3) projections (WMMA GEMMs)
  const dim3 gemmGrid(8, 64);
  gemm512<1><<<gemmGrid, 128, 0, stream>>>(Xh, Wtq, q_b, nullptr, Qf);
  gemm512<1><<<gemmGrid, 128, 0, stream>>>(Yh, Wtk, k_b, nullptr, Kf);
  gemm512<2><<<gemmGrid, 128, 0, stream>>>(Yh, Wtv, v_b, nullptr, Vtr);

  // 4) mask thresholds from K row-sums
  ksum_kernel<<<64, 256, 0, stream>>>(Kf, Ksum);
  thre_kernel<<<dim3(4, 64), 256, 0, stream>>>(Qf, Ksum, Thre);

  // 5) fused attention (WMMA)
  attn_kernel<<<dim3(8, 64), 128, 0, stream>>>(Qf, Kf, Vtr, Thre, Of);

  // 6) output projection (WMMA) -> f32 d_out
  gemm512<0><<<gemmGrid, 128, 0, stream>>>(Of, Wto, o_b, out, nullptr);
}
